// DipNet_47467978555504
// MI455X (gfx1250) — compile-verified, matrix operand order checked by reading, not run on hardware
//
#include <hip/hip_runtime.h>
#include <hip/hip_bf16.h>

typedef __attribute__((ext_vector_type(16))) _Float16 v16h;
typedef __attribute__((ext_vector_type(8)))  _Float16 v8h;
typedef __attribute__((ext_vector_type(8)))  float    v8f;

#define B_   512
#define S_   17
#define V_   13042
#define L_   81
#define NB_  16
#define E_   120
#define P_   60
#define SEc_ 20
#define OE_  80
#define H_   200
#define BS_  35
#define PO_  40
#define MOD_ 80
#define XIN_ 320
#define G4H_ 800
#define EPS_ 1e-5f

#define LDK_ 40   // padded K stride in halves: 80 bytes, 16B aligned, conflict-friendly

// ---------------------------------------------------------------------------
// Generic batched GEMM: C[z] = A[z][M,K] * B[z][K,N] (+bias[N])
// Block tile 32(M) x 64(N), K chunked by 32. 128 threads = 4 waves,
// each wave computes 16x32 with two v_wmma_f32_16x16x32_f16 sharing one
// A fragment. f32 operands converted to f16 during LDS staging; f32 accum.
// bTrans=1 means B is stored [N,K] (compute A @ B.T).
// ---------------------------------------------------------------------------
__launch_bounds__(128)
__global__ void gemm_wmma(const float* __restrict__ A, long long lda, long long Abs,
                          const float* __restrict__ Bm, long long ldb, long long Bbs, int bTrans,
                          const float* __restrict__ bias,
                          float* __restrict__ C, long long ldc, long long Cbs,
                          int M, int N, int K)
{
    __shared__ alignas(16) _Float16 ash[2][32][LDK_];
    __shared__ alignas(16) _Float16 bsh[2][64][LDK_];

    const int z = blockIdx.z;
    const float* Ab = A + (long long)z * Abs;
    const float* Bb = Bm + (long long)z * Bbs;
    float* Cb = C + (long long)z * Cbs;

    const int tileN = blockIdx.x * 64;
    const int tileM = blockIdx.y * 32;

    const int tid  = threadIdx.x;      // 0..127
    const int lane = tid & 31;
    const int wave = tid >> 5;         // 0..3
    const int wm   = (wave >> 1) * 16; // 0 / 16
    const int wn   = (wave & 1) * 32;  // 0 / 32

    v8f acc0 = {};
    v8f acc1 = {};

    const int nChunks = (K + 31) >> 5;

    auto stageA = [&](int k0, _Float16 (*as)[LDK_]) {
        if (tileM + 32 <= M && k0 + 32 <= K) {
            // fast path: 256 groups of 4 contiguous k, unconditional loads
            for (int i = tid; i < 256; i += 128) {
                int m = i >> 3, c = (i & 7) << 2;
                const float* p = Ab + (long long)(tileM + m) * lda + (k0 + c);
                float r0 = p[0], r1 = p[1], r2 = p[2], r3 = p[3];
                as[m][c + 0] = (_Float16)r0;
                as[m][c + 1] = (_Float16)r1;
                as[m][c + 2] = (_Float16)r2;
                as[m][c + 3] = (_Float16)r3;
            }
        } else {
            for (int i = tid; i < 1024; i += 128) {
                int m = i >> 5, k = i & 31;
                int gm = tileM + m, gk = k0 + k;
                float v = (gm < M && gk < K) ? Ab[(long long)gm * lda + gk] : 0.0f;
                as[m][k] = (_Float16)v;
            }
        }
    };

    auto stageB = [&](int k0, _Float16 (*bs)[LDK_]) {
        if (bTrans) {
            if (tileN + 64 <= N && k0 + 32 <= K) {
                for (int i = tid; i < 512; i += 128) {
                    int n = i >> 3, c = (i & 7) << 2;
                    const float* p = Bb + (long long)(tileN + n) * ldb + (k0 + c);
                    float r0 = p[0], r1 = p[1], r2 = p[2], r3 = p[3];
                    bs[n][c + 0] = (_Float16)r0;
                    bs[n][c + 1] = (_Float16)r1;
                    bs[n][c + 2] = (_Float16)r2;
                    bs[n][c + 3] = (_Float16)r3;
                }
            } else {
                for (int i = tid; i < 2048; i += 128) {
                    int n = i >> 5, k = i & 31;
                    int gn = tileN + n, gk = k0 + k;
                    float v = (gn < N && gk < K) ? Bb[(long long)gn * ldb + gk] : 0.0f;
                    bs[n][k] = (_Float16)v;
                }
            }
        } else {
            // B stored [K, N]: read coalesced over n, write transposed to LDS
            if (tileN + 64 <= N && k0 + 32 <= K) {
                for (int i = tid; i < 512; i += 128) {
                    int k = i >> 4, n4 = (i & 15) << 2;
                    const float* p = Bb + (long long)(k0 + k) * ldb + (tileN + n4);
                    float r0 = p[0], r1 = p[1], r2 = p[2], r3 = p[3];
                    bs[n4 + 0][k] = (_Float16)r0;
                    bs[n4 + 1][k] = (_Float16)r1;
                    bs[n4 + 2][k] = (_Float16)r2;
                    bs[n4 + 3][k] = (_Float16)r3;
                }
            } else {
                for (int i = tid; i < 2048; i += 128) {
                    int k = i >> 6, n = i & 63;
                    int gn = tileN + n, gk = k0 + k;
                    float v = (gn < N && gk < K) ? Bb[(long long)gk * ldb + gn] : 0.0f;
                    bs[n][k] = (_Float16)v;
                }
            }
        }
    };

    stageA(0, ash[0]);
    stageB(0, bsh[0]);
    __syncthreads();

    const int half = lane >> 4;   // K-half select per documented layout
    const int l16  = lane & 15;

    for (int ch = 0; ch < nChunks; ++ch) {
        const int cur = ch & 1;
        if (ch + 1 < nChunks) {
            stageA((ch + 1) << 5, ash[cur ^ 1]);
            stageB((ch + 1) << 5, bsh[cur ^ 1]);
        }
        // prefetch two chunks ahead into cache (global_prefetch_b8)
        if (ch + 2 < nChunks && tid < 32) {
            int gm = tileM + tid;
            int gk = (ch + 2) << 5;
            if (gm < M) __builtin_prefetch(&Ab[(long long)gm * lda + gk], 0, 1);
        }

        // A fragment: two contiguous 16-byte runs -> ds_load_b128 pair
        const _Float16* ar = &ash[cur][wm + l16][0];
        v8h a0 = *(const v8h*)(ar + 8 * half);
        v8h a1 = *(const v8h*)(ar + 16 + 8 * half);
        v16h af = __builtin_shufflevector(a0, a1, 0, 1, 2, 3, 4, 5, 6, 7,
                                          8, 9, 10, 11, 12, 13, 14, 15);
        // B fragments for the two 16-wide n-subtiles
        const _Float16* br0 = &bsh[cur][wn + l16][0];
        v8h b00 = *(const v8h*)(br0 + 8 * half);
        v8h b01 = *(const v8h*)(br0 + 16 + 8 * half);
        v16h bf0 = __builtin_shufflevector(b00, b01, 0, 1, 2, 3, 4, 5, 6, 7,
                                           8, 9, 10, 11, 12, 13, 14, 15);
        const _Float16* br1 = &bsh[cur][wn + 16 + l16][0];
        v8h b10 = *(const v8h*)(br1 + 8 * half);
        v8h b11 = *(const v8h*)(br1 + 16 + 8 * half);
        v16h bf1 = __builtin_shufflevector(b10, b11, 0, 1, 2, 3, 4, 5, 6, 7,
                                           8, 9, 10, 11, 12, 13, 14, 15);

        acc0 = __builtin_amdgcn_wmma_f32_16x16x32_f16(false, af, false, bf0,
                                                      (short)0, acc0, false, false);
        acc1 = __builtin_amdgcn_wmma_f32_16x16x32_f16(false, af, false, bf1,
                                                      (short)0, acc1, false, false);
        __syncthreads();
    }

    // C write per 32-bit 16x16 C/D layout: VGPR r -> row r + 8*(lane/16)
    const int col0 = tileN + wn + l16;
    const int col1 = col0 + 16;
#pragma unroll
    for (int r = 0; r < 8; ++r) {
        int row = tileM + wm + r + 8 * half;
        if (row < M) {
            if (col0 < N) {
                float v = acc0[r];
                if (bias) v += bias[col0];
                Cb[(long long)row * ldc + col0] = v;
            }
            if (col1 < N) {
                float v = acc1[r];
                if (bias) v += bias[col1];
                Cb[(long long)row * ldc + col1] = v;
            }
        }
    }
}

// ---------------------------------------------------------------------------
// mod = concat(power_emb, season_emb)   [B, 80]
// ---------------------------------------------------------------------------
__global__ void mod_kernel(const float* __restrict__ p1h, const float* __restrict__ s1h,
                           const float* __restrict__ pW, const float* __restrict__ pb,
                           const float* __restrict__ sW, const float* __restrict__ sb,
                           float* __restrict__ modo)
{
    int i = blockIdx.x * blockDim.x + threadIdx.x;
    if (i >= B_ * MOD_) return;
    int b = i / MOD_, j = i % MOD_;
    float v;
    if (j < P_) {
        v = pb[j];
        for (int k = 0; k < 7; ++k) v += p1h[b * 7 + k] * pW[j * 7 + k];
    } else {
        int q = j - P_;
        v = sb[q];
        for (int k = 0; k < 3; ++k) v += s1h[b * 3 + k] * sW[q * 3 + k];
    }
    modo[i] = v;
}

// ---------------------------------------------------------------------------
// BatchNorm stats per location l: mean/var over (B, E)
// ---------------------------------------------------------------------------
__global__ void bn_stats(const float* __restrict__ Y, float* __restrict__ mu,
                         float* __restrict__ rs)
{
    int l = blockIdx.x;
    __shared__ float s1[256], s2[256];
    float a = 0.f, q = 0.f;
    for (int i = threadIdx.x; i < B_ * E_; i += 256) {
        int bb = i / E_, o = i % E_;
        float v = Y[(long long)bb * L_ * E_ + (long long)l * E_ + o];
        a += v; q += v * v;
    }
    s1[threadIdx.x] = a; s2[threadIdx.x] = q;
    __syncthreads();
    for (int st = 128; st > 0; st >>= 1) {
        if (threadIdx.x < st) {
            s1[threadIdx.x] += s1[threadIdx.x + st];
            s2[threadIdx.x] += s2[threadIdx.x + st];
        }
        __syncthreads();
    }
    if (threadIdx.x == 0) {
        float m = s1[0] / (float)(B_ * E_);
        float var = s2[0] / (float)(B_ * E_) - m * m;
        mu[l] = m;
        rs[l] = rsqrtf(var + EPS_);
    }
}

// ---------------------------------------------------------------------------
// FiLM: out = relu(gamma * bn(y2) + beta) [+ x residual], strided output
// ---------------------------------------------------------------------------
__global__ void film_apply(const float* __restrict__ Y2, const float* __restrict__ Xin,
                           const float* __restrict__ mu, const float* __restrict__ rs,
                           const float* __restrict__ bnw, const float* __restrict__ bnb,
                           const float* __restrict__ gamma, const float* __restrict__ beta,
                           int residual, float* __restrict__ Out, int outLd, int outOff)
{
    long long i = (long long)blockIdx.x * blockDim.x + threadIdx.x;
    if (i >= (long long)B_ * L_ * E_) return;
    int o = (int)(i % E_);
    long long t = i / E_;
    int l = (int)(t % L_);
    int b = (int)(t / L_);
    float y = Y2[i];
    y = (y - mu[l]) * rs[l] * bnw[l] + bnb[l];
    float v = gamma[b * E_ + o] * y + beta[b * E_ + o];
    v = fmaxf(v, 0.0f);
    if (residual) v += Xin[i];
    Out[((long long)b * L_ + l) * outLd + outOff + o] = v;
}

// ---------------------------------------------------------------------------
// invalid[b] = !any(loc_idxs[b, :] != -1)
// ---------------------------------------------------------------------------
__global__ void invalid_kernel(const int* __restrict__ loc, int* __restrict__ invalid)
{
    int b = blockIdx.x;
    __shared__ int any;
    if (threadIdx.x == 0) any = 0;
    __syncthreads();
    for (int l = threadIdx.x; l < L_; l += blockDim.x)
        if (loc[b * L_ + l] != -1) atomicOr(&any, 1);
    __syncthreads();
    if (threadIdx.x == 0) invalid[b] = (any == 0);
}

// ---------------------------------------------------------------------------
// init LSTM state + cumulative order mask
// ---------------------------------------------------------------------------
__global__ void init_state(float* __restrict__ h, float* __restrict__ c,
                           float* __restrict__ oemb, unsigned char* __restrict__ cum)
{
    long long i = (long long)blockIdx.x * blockDim.x + threadIdx.x;
    if (i < (long long)B_ * V_) cum[i] = 1;
    if (i < B_ * H_) { h[i] = 0.f; c[i] = 0.f; }
    if (i < B_ * OE_) oemb[i] = 0.f;
}

// ---------------------------------------------------------------------------
// Alignment weights for one step
// ---------------------------------------------------------------------------
__global__ void ali_kernel(const int* __restrict__ loc, const float* __restrict__ MA,
                           int step, float* __restrict__ ali)
{
    int b = blockIdx.x;
    __shared__ float sel[L_];
    __shared__ float raw[L_];
    __shared__ float ssum;
    for (int l = threadIdx.x; l < L_; l += blockDim.x) {
        int li = loc[b * L_ + l];
        sel[l] = (li == step || li == -2) ? 1.0f : 0.0f;
    }
    __syncthreads();
    for (int n = threadIdx.x; n < L_; n += blockDim.x) {
        float a = 0.f;
        for (int l = 0; l < L_; ++l) a += sel[l] * MA[l * L_ + n];
        raw[n] = a;
    }
    __syncthreads();
    if (threadIdx.x == 0) {
        float s = 0.f;
        for (int n = 0; n < L_; ++n) s += raw[n];
        ssum = s;
    }
    __syncthreads();
    float s = ssum;
    for (int n = threadIdx.x; n < L_; n += blockDim.x)
        ali[b * L_ + n] = (s > 0.f) ? raw[n] / s : 0.f;
}

// ---------------------------------------------------------------------------
// xin = concat(ali @ enc, oemb)   [B, 320]
// ---------------------------------------------------------------------------
__global__ void xin_kernel(const float* __restrict__ ali, const float* __restrict__ enc,
                           const float* __restrict__ oemb, float* __restrict__ xin)
{
    int b = blockIdx.x;
    int d = threadIdx.x;  // blockDim.x == 320
    if (d < 2 * E_) {
        float a = 0.f;
        for (int l = 0; l < L_; ++l)
            a += ali[b * L_ + l] * enc[((long long)b * L_ + l) * (2 * E_) + d];
        xin[b * XIN_ + d] = a;
    } else {
        xin[b * XIN_ + d] = oemb[b * OE_ + (d - 2 * E_)];
    }
}

// ---------------------------------------------------------------------------
// LSTM pointwise (torch gate order i,f,g,o)
// ---------------------------------------------------------------------------
__global__ void lstm_kernel(const float* __restrict__ g1, const float* __restrict__ g2,
                            const float* __restrict__ b_ih, const float* __restrict__ b_hh,
                            float* __restrict__ h, float* __restrict__ c)
{
    int i = blockIdx.x * blockDim.x + threadIdx.x;
    if (i >= B_ * H_) return;
    int b = i / H_, j = i % H_;
    const float* G1 = g1 + b * G4H_;
    const float* G2 = g2 + b * G4H_;
    float gi = G1[j]          + G2[j]          + b_ih[j]          + b_hh[j];
    float gf = G1[H_ + j]     + G2[H_ + j]     + b_ih[H_ + j]     + b_hh[H_ + j];
    float gg = G1[2 * H_ + j] + G2[2 * H_ + j] + b_ih[2 * H_ + j] + b_hh[2 * H_ + j];
    float go = G1[3 * H_ + j] + G2[3 * H_ + j] + b_ih[3 * H_ + j] + b_hh[3 * H_ + j];
    float sf = 1.0f / (1.0f + __expf(-gf));
    float si = 1.0f / (1.0f + __expf(-gi));
    float so = 1.0f / (1.0f + __expf(-go));
    float cn = sf * c[i] + si * tanhf(gg);
    c[i] = cn;
    h[i] = so * tanhf(cn);
}

// ---------------------------------------------------------------------------
// Mask scores, add Gumbel noise (hash RNG), argmax, update oemb/cum
// ---------------------------------------------------------------------------
__device__ __forceinline__ float u01_hash(unsigned int x)
{
    x ^= x >> 17; x *= 0xed5ad4bbu;
    x ^= x >> 11; x *= 0xac4c1b51u;
    x ^= x >> 15; x *= 0x31848babu;
    x ^= x >> 14;
    return ((float)(x >> 8) + 0.5f) * (1.0f / 16777216.0f);
}

__global__ void sample_kernel(float* __restrict__ scores, const unsigned char* __restrict__ masks,
                              unsigned char* __restrict__ cum, const int* __restrict__ invalid,
                              const float* __restrict__ emb_table, float* __restrict__ oemb,
                              float* __restrict__ out_idx, int step)
{
    int b = blockIdx.x;
    float* sc = scores + ((long long)b * S_ + step) * V_;
    const unsigned char* mk = masks + ((long long)b * S_ + step) * V_;
    unsigned char* cm = cum + (long long)b * V_;
    int inv = invalid[b];

    float best = -1e30f;
    int bidx = 0;
    for (int v = threadIdx.x; v < V_; v += blockDim.x) {
        int m = inv ? 1 : ((mk[v] != 0) && (cm[v] != 0));
        float s = sc[v];
        float cap = m ? 9e8f : -1e8f;     // m*1e9 - 1e8
        s = fminf(s, cap);
        sc[v] = s;                        // masked scores are the returned scores
        unsigned int hh = (unsigned int)step * 0x9E3779B9u
                        ^ (unsigned int)b    * 0x85EBCA6Bu
                        ^ (unsigned int)v    * 0xC2B2AE35u
                        ^ 0x6A09E667u;
        float u = u01_hash(hh);
        float g = -__logf(-__logf(u));
        float val = s + g;                // temperature = 1
        if (val > best) { best = val; bidx = v; }
    }
    __shared__ float sv[256];
    __shared__ int   six[256];
    sv[threadIdx.x] = best; six[threadIdx.x] = bidx;
    __syncthreads();
    for (int st = 128; st > 0; st >>= 1) {
        if (threadIdx.x < st) {
            if (sv[threadIdx.x + st] > sv[threadIdx.x]) {
                sv[threadIdx.x]  = sv[threadIdx.x + st];
                six[threadIdx.x] = six[threadIdx.x + st];
            }
        }
        __syncthreads();
    }
    int idx = six[0];
    if (threadIdx.x == 0) {
        out_idx[b * S_ + step] = (float)idx;
        cm[idx] = 0;
    }
    if (threadIdx.x < OE_)
        oemb[b * OE_ + threadIdx.x] = emb_table[(long long)idx * OE_ + threadIdx.x];
}

// ---------------------------------------------------------------------------
extern "C" void kernel_launch(void* const* d_in, const int* in_sizes, int n_in,
                              void* d_out, int out_size, void* d_ws, size_t ws_size,
                              hipStream_t stream)
{
    (void)in_sizes; (void)n_in; (void)out_size; (void)ws_size;

    const float* x_bo     = (const float*)d_in[0];
    const float* x_po     = (const float*)d_in[1];
    const float* x_power  = (const float*)d_in[2];
    const float* x_season = (const float*)d_in[3];
    /* d_in[4] in_adj_phase: unused by the forward pass */
    const int*   loc_idxs = (const int*)d_in[5];
    const unsigned char* order_masks = (const unsigned char*)d_in[6];
    const float* power_W  = (const float*)d_in[7];
    const float* power_b  = (const float*)d_in[8];
    const float* season_W = (const float*)d_in[9];
    const float* season_b = (const float*)d_in[10];
    const float* Adj      = (const float*)d_in[11];

    const float* gcW0[2] = {(const float*)d_in[12], (const float*)d_in[21]};
    const float* gcWr[2] = {(const float*)d_in[13], (const float*)d_in[22]};
    const float* gcb [2] = {(const float*)d_in[14], (const float*)d_in[23]};
    const float* bnw [2] = {(const float*)d_in[15], (const float*)d_in[24]};
    const float* bnb [2] = {(const float*)d_in[16], (const float*)d_in[25]};
    const float* fgW [2] = {(const float*)d_in[17], (const float*)d_in[26]};
    const float* fgb [2] = {(const float*)d_in[18], (const float*)d_in[27]};
    const float* fbW [2] = {(const float*)d_in[19], (const float*)d_in[28]};
    const float* fbb [2] = {(const float*)d_in[20], (const float*)d_in[29]};
    const float* x0  [2] = {x_bo, x_po};
    const int    K0  [2] = {BS_, PO_};

    const float* order_emb = (const float*)d_in[30];
    const float* W_ih = (const float*)d_in[31];
    const float* W_hh = (const float*)d_in[32];
    const float* b_ih = (const float*)d_in[33];
    const float* b_hh = (const float*)d_in[34];
    const float* out_W = (const float*)d_in[35];
    const float* out_b = (const float*)d_in[36];
    const float* MA = (const float*)d_in[37];

    // ---- workspace carve-out ----
    float* W = (float*)d_ws;
    size_t off = 0;
    auto alloc = [&](size_t n) { float* p = W + off; off += n; return p; };
    float* modb   = alloc((size_t)B_ * MOD_);
    float* gammab = alloc((size_t)B_ * E_);
    float* betab  = alloc((size_t)B_ * E_);
    float* mub    = alloc(128);
    float* rsb    = alloc(128);
    float* bufA   = alloc((size_t)B_ * L_ * E_);
    float* bufY   = alloc((size_t)B_ * L_ * E_);
    float* bufZ   = alloc((size_t)B_ * L_ * E_);
    float* encb   = alloc((size_t)B_ * L_ * 2 * E_);
    float* hb     = alloc((size_t)B_ * H_);
    float* cb     = alloc((size_t)B_ * H_);
    float* oembb  = alloc((size_t)B_ * OE_);
    float* alib   = alloc((size_t)B_ * L_);
    float* xinb   = alloc((size_t)B_ * XIN_);
    float* g1b    = alloc((size_t)B_ * G4H_);
    float* g2b    = alloc((size_t)B_ * G4H_);
    int*   invb   = (int*)alloc(B_);
    unsigned char* cumb = (unsigned char*)(W + off);

    float* outIdx     = (float*)d_out;
    float* scoresBase = (float*)d_out + (size_t)B_ * S_;

    // ---- embeddings / flags / state init ----
    mod_kernel<<<(B_ * MOD_ + 255) / 256, 256, 0, stream>>>(
        x_power, x_season, power_W, power_b, season_W, season_b, modb);
    invalid_kernel<<<B_, 96, 0, stream>>>(loc_idxs, invb);
    {
        long long tot = (long long)B_ * V_;
        init_state<<<(unsigned)((tot + 255) / 256), 256, 0, stream>>>(hb, cb, oembb, cumb);
    }

    // ---- GC/FiLM trunk: 2 streams x 16 blocks ----
    const int filmBlocks = (int)(((long long)B_ * L_ * E_ + 255) / 256);
    for (int st = 0; st < 2; ++st) {
        for (int blk = 0; blk < NB_; ++blk) {
            const int Kin = (blk == 0) ? K0[st] : E_;
            const float* Aptr = (blk == 0) ? x0[st] : bufA;
            const long long lda = (long long)L_ * Kin;
            const long long Abs = Kin;
            const float* Bptr = (blk == 0) ? gcW0[st]
                                           : gcWr[st] + (long long)(blk - 1) * L_ * E_ * E_;
            const long long Bbs = (long long)Kin * E_;

            // 1) per-location graph-conv: Y[b,l,:] = X[b,l,:] @ W[l]
            gemm_wmma<<<dim3(2, 16, L_), 128, 0, stream>>>(
                Aptr, lda, Abs, Bptr, E_, Bbs, /*bTrans=*/0, nullptr,
                bufY, (long long)L_ * E_, E_, B_, E_, Kin);

            // 2) adjacency mix per batch item: Z[b] = Adj @ Y[b]  (+ gc bias)
            gemm_wmma<<<dim3(2, 3, B_), 128, 0, stream>>>(
                Adj, L_, 0, bufY, E_, (long long)L_ * E_, /*bTrans=*/0,
                gcb[st] + blk * E_,
                bufZ, E_, (long long)L_ * E_, L_, E_, L_);

            // 3) batch-norm stats per location
            bn_stats<<<L_, 256, 0, stream>>>(bufZ, mub, rsb);

            // 4) FiLM gamma/beta: mod @ W.T + b
            gemm_wmma<<<dim3(2, 16, 1), 128, 0, stream>>>(
                modb, MOD_, 0, fgW[st] + (long long)blk * E_ * MOD_, MOD_, 0, /*bTrans=*/1,
                fgb[st] + blk * E_, gammab, E_, 0, B_, E_, MOD_);
            gemm_wmma<<<dim3(2, 16, 1), 128, 0, stream>>>(
                modb, MOD_, 0, fbW[st] + (long long)blk * E_ * MOD_, MOD_, 0, /*bTrans=*/1,
                fbb[st] + blk * E_, betab, E_, 0, B_, E_, MOD_);

            // 5) BN + FiLM + relu (+ residual); final block writes into enc
            const int residual = (blk > 0) ? 1 : 0;
            float* outP = (blk == NB_ - 1) ? encb : bufA;
            const int outLd  = (blk == NB_ - 1) ? 2 * E_ : E_;
            const int outOff = (blk == NB_ - 1) ? st * E_ : 0;
            film_apply<<<filmBlocks, 256, 0, stream>>>(
                bufZ, bufA, mub, rsb, bnw[st] + blk * L_, bnb[st] + blk * L_,
                gammab, betab, residual, outP, outLd, outOff);
        }
    }

    // ---- LSTM decoder scan ----
    for (int s = 0; s < S_; ++s) {
        ali_kernel<<<B_, 128, 0, stream>>>(loc_idxs, MA, s, alib);
        xin_kernel<<<B_, XIN_, 0, stream>>>(alib, encb, oembb, xinb);

        // gates = xin @ W_ih.T ; h @ W_hh.T
        gemm_wmma<<<dim3(13, 16, 1), 128, 0, stream>>>(
            xinb, XIN_, 0, W_ih, XIN_, 0, /*bTrans=*/1, nullptr,
            g1b, G4H_, 0, B_, G4H_, XIN_);
        gemm_wmma<<<dim3(13, 16, 1), 128, 0, stream>>>(
            hb, H_, 0, W_hh, H_, 0, /*bTrans=*/1, nullptr,
            g2b, G4H_, 0, B_, G4H_, H_);

        lstm_kernel<<<(B_ * H_ + 255) / 256, 256, 0, stream>>>(
            g1b, g2b, b_ih, b_hh, hb, cb);

        // scores[:, s, :] = h @ out_W.T + out_b  (written strided into d_out)
        gemm_wmma<<<dim3((V_ + 63) / 64, 16, 1), 128, 0, stream>>>(
            hb, H_, 0, out_W, H_, 0, /*bTrans=*/1, out_b,
            scoresBase + (long long)s * V_, (long long)S_ * V_, 0, B_, V_, H_);

        sample_kernel<<<B_, 256, 0, stream>>>(
            scoresBase, order_masks, cumb, invb, order_emb, oembb, outIdx, s);
    }
}
